// Encoder_84069689852135
// MI455X (gfx1250) — matrix operand order repaired and checked by reading, not compiled, not run on hardware
//
#include <hip/hip_runtime.h>

#define D_DIM 64
#define H_DIM 128
#define KIN   8
#define KOUT  4

typedef __bf16 bf16x16 __attribute__((ext_vector_type(16)));
typedef float  f32x8   __attribute__((ext_vector_type(8)));
typedef float  f32x4   __attribute__((ext_vector_type(4)));

union ARegs { unsigned int u[8]; bf16x16 v; };

// round-to-nearest-even bf16 (result in bits [31:16])
__device__ __forceinline__ unsigned int bf_rnd(float f) {
  unsigned int u = __float_as_uint(f);
  return u + 0x7FFFu + ((u >> 16) & 1u);
}
// pack two floats into {bf16(hi), bf16(lo)} with one v_perm_b32
__device__ __forceinline__ unsigned int pack_bf2(float lo, float hi) {
  return __builtin_amdgcn_perm(bf_rnd(hi), bf_rnd(lo), 0x07060302u);
}
// Wave-local LDS ordering: DS ops from one wave execute in order at the LDS
// unit; wait for outstanding DS ops and stop compiler reordering.
__device__ __forceinline__ void lds_fence() {
  asm volatile("s_wait_dscnt 0" ::: "memory");
  __builtin_amdgcn_wave_barrier();
}

#define EIN_STRIDE 68   // pad 64->68 floats: distinct banks per edge row
#define PO_STRIDE  72   // pad 64->72 floats
#define H_STRIDE   136  // pad 128->136 halfwords: distinct banks per row

struct WaveLds {
  float ein[KIN * EIN_STRIDE];     // 2176 B
  float pooled[4 * PO_STRIDE];     // 1152 B
  float ws[8];                     //   32 B
  unsigned short h[16 * H_STRIDE]; // 4352 B (bf16 bits)
};                                 // 7712 B per wave (16B aligned)

// pack 8 contiguous floats at p into A.u[base .. base+3]
__device__ __forceinline__ void pack8(ARegs& A, int base, const float* p) {
  f32x4 f0 = *(const f32x4*)p;
  f32x4 f1 = *(const f32x4*)(p + 4);
  A.u[base + 0] = pack_bf2(f0[0], f0[1]);
  A.u[base + 1] = pack_bf2(f0[2], f0[3]);
  A.u[base + 2] = pack_bf2(f1[0], f1[1]);
  A.u[base + 3] = pack_bf2(f1[2], f1[3]);
}

__global__ __launch_bounds__(256) void encoder_kernel(
    const float* __restrict__ edges,
    const int*   __restrict__ in_idx,
    const int*   __restrict__ in_mask,
    const int*   __restrict__ out_idx,
    const int*   __restrict__ out_mask,
    const float* __restrict__ b0,
    const float* __restrict__ b1,
    const float* __restrict__ Wout,
    const float* __restrict__ bout,
    const unsigned short* __restrict__ Wb,   // [2][128][128] bf16 (W0 ++ W1)
    float* __restrict__ out,
    int N)
{
  __shared__ WaveLds lds[8];
  const int lane = threadIdx.x & 31;
  const int wave = threadIdx.x >> 5;
  WaveLds& L = lds[wave];

  const int nodeBase = blockIdx.x * 32 + wave * 4;   // 4 nodes per wave
  if (nodeBase >= N) return;                          // wave-uniform

  // ======================= attention -> pooled[4][64] ======================
  const int er = lane >> 2;          // edge row 0..7 for gather
  const int ec = (lane & 3) * 16;    // 16-float slice
  const int q  = lane >> 2;          // query 0..7
  const int kk = lane & 3;           // key pair (kk, kk+4)

  for (int nl = 0; nl < 4; ++nl) {
    const int n = nodeBase + nl;
    // gather Ein row -> LDS
    {
      const long eidx = in_idx[n * KIN + er];
      const f32x4* src = (const f32x4*)(edges + eidx * D_DIM + ec);
      f32x4* dst = (f32x4*)&L.ein[er * EIN_STRIDE + ec];
      dst[0] = src[0]; dst[1] = src[1]; dst[2] = src[2]; dst[3] = src[3];
    }
    lds_fence();
    // scores s[q][kk], s[q][kk+4] -- packed f32 math (v_pk_fma_f32)
    f32x4 v0 = (f32x4)0.f, v1 = (f32x4)0.f;
    #pragma unroll
    for (int d = 0; d < D_DIM; d += 4) {
      f32x4 a = *(const f32x4*)&L.ein[q * EIN_STRIDE + d];
      f32x4 x = *(const f32x4*)&L.ein[kk * EIN_STRIDE + d];
      f32x4 y = *(const f32x4*)&L.ein[(kk + 4) * EIN_STRIDE + d];
      v0 += a * x;
      v1 += a * y;
    }
    float s0 = ((v0[0] + v0[2]) + (v0[1] + v0[3])) * 0.125f;  // 1/sqrt(64)
    float s1 = ((v1[0] + v1[2]) + (v1[1] + v1[3])) * 0.125f;
    const int mk0 = in_mask[n * KIN + kk];
    const int mk1 = in_mask[n * KIN + kk + 4];
    s0 = mk0 ? s0 : -1e9f;
    s1 = mk1 ? s1 : -1e9f;
    // softmax over 8 keys (4 lanes x 2 regs within the q-group)
    float mx = fmaxf(s0, s1);
    mx = fmaxf(mx, __shfl_xor(mx, 1, 32));
    mx = fmaxf(mx, __shfl_xor(mx, 2, 32));
    float e0 = __expf(s0 - mx), e1 = __expf(s1 - mx);
    float sm = e0 + e1;
    sm += __shfl_xor(sm, 1, 32);
    sm += __shfl_xor(sm, 2, 32);
    const float a0 = e0 / sm, a1 = e1 / sm;
    // per-key pooling weights: w[k] = sum_q m_q * attn[q][k] / denom
    const float fq = (in_mask[n * KIN + q] != 0) ? 1.f : 0.f;
    float w0 = fq * a0, w1 = fq * a1, dn = fq;
    #pragma unroll
    for (int m = 4; m <= 16; m <<= 1) {
      w0 += __shfl_xor(w0, m, 32);
      w1 += __shfl_xor(w1, m, 32);
      dn += __shfl_xor(dn, m, 32);
    }
    dn = fmaxf(dn, 1.f);
    w0 /= dn; w1 /= dn;
    if (lane < 4) { L.ws[lane] = w0; L.ws[lane + 4] = w1; }
    lds_fence();
    // pooled[d] = sum_k w[k] * Ein[k][d]   (lane covers d and d+32)
    float p0 = 0.f, p1 = 0.f;
    #pragma unroll
    for (int k = 0; k < 8; ++k) {
      const float wk = L.ws[k];
      p0 += wk * L.ein[k * EIN_STRIDE + lane];
      p1 += wk * L.ein[k * EIN_STRIDE + lane + 32];
    }
    L.pooled[nl * PO_STRIDE + lane]      = p0;
    L.pooled[nl * PO_STRIDE + lane + 32] = p1;
    lds_fence();
  }

  // ============================ MLP (WMMA) =================================
  const int half = lane >> 4;   // K-half select per ISA 16-bit A/B layouts
  const int mrow = lane & 15;   // M row / N col
  const int nl2  = mrow >> 2;   // node-local
  const int ko   = mrow & 3;    // out-edge
  const long eo  = out_idx[(nodeBase + nl2) * KOUT + ko];
  const float* xrow = edges + eo * D_DIM;

  // A0: x = [edge(64) ++ pooled(64)] in 16-bit A layout (K groups of 8/half)
  ARegs A0[4];
  #pragma unroll
  for (int c = 0; c < 2; ++c) {        // K 0..63 from gathered edge row
    const int k0 = c * 32 + half * 8;
    pack8(A0[c], 0, xrow + k0);
    pack8(A0[c], 4, xrow + k0 + 16);
  }
  #pragma unroll
  for (int c = 0; c < 2; ++c) {        // K 64..127 from pooled (LDS)
    const int k0 = c * 32 + half * 8;
    const float* pr = &L.pooled[nl2 * PO_STRIDE + k0];
    pack8(A0[2 + c], 0, pr);
    pack8(A0[2 + c], 4, pr + 16);
  }

  // ---- layer 0: h = relu(x @ W0^T + b0), h -> LDS (bf16) ----
  #pragma unroll
  for (int t = 0; t < 8; ++t) {
    const int oc = t * 16 + mrow;
    f32x8 acc;
    const float bias = b0[oc];
    #pragma unroll
    for (int r = 0; r < 8; ++r) acc[r] = bias;
    #pragma unroll
    for (int c = 0; c < 4; ++c) {
      ARegs B;   // B layout: K groups of 8 per half (k = half*8 + i, 16 + half*8 + i)
      const uint4* bp = (const uint4*)(Wb + oc * H_DIM + c * 32 + half * 8);
      uint4 u0 = bp[0], u1 = bp[2];
      B.u[0] = u0.x; B.u[1] = u0.y; B.u[2] = u0.z; B.u[3] = u0.w;
      B.u[4] = u1.x; B.u[5] = u1.y; B.u[6] = u1.z; B.u[7] = u1.w;
      acc = __builtin_amdgcn_wmma_f32_16x16x32_bf16(
          false, A0[c].v, false, B.v, (short)0, acc, false, false);
    }
    #pragma unroll
    for (int r = 0; r < 8; ++r) {
      const float v = fmaxf(acc[r], 0.f);
      L.h[(r + 8 * half) * H_STRIDE + oc] = (unsigned short)(bf_rnd(v) >> 16);
    }
  }
  lds_fence();

  // A1 from h (already bf16, single uint4 DS loads per K-group)
  ARegs A1[4];
  #pragma unroll
  for (int c = 0; c < 4; ++c) {
    const int k0 = c * 32 + half * 8;
    uint4 u0 = *(const uint4*)&L.h[mrow * H_STRIDE + k0];
    uint4 u1 = *(const uint4*)&L.h[mrow * H_STRIDE + k0 + 16];
    A1[c].u[0] = u0.x; A1[c].u[1] = u0.y; A1[c].u[2] = u0.z; A1[c].u[3] = u0.w;
    A1[c].u[4] = u1.x; A1[c].u[5] = u1.y; A1[c].u[6] = u1.z; A1[c].u[7] = u1.w;
  }

  // ---- layer 1 + fused output dot: y = relu(h@W1^T+b1) @ Wout + bout ----
  float accout[8];
  #pragma unroll
  for (int r = 0; r < 8; ++r) accout[r] = 0.f;
  const unsigned short* Wb1 = Wb + H_DIM * H_DIM;
  #pragma unroll
  for (int t = 0; t < 8; ++t) {
    const int oc = t * 16 + mrow;
    f32x8 acc;
    const float bias = b1[oc];
    #pragma unroll
    for (int r = 0; r < 8; ++r) acc[r] = bias;
    #pragma unroll
    for (int c = 0; c < 4; ++c) {
      ARegs B;
      const uint4* bp = (const uint4*)(Wb1 + oc * H_DIM + c * 32 + half * 8);
      uint4 u0 = bp[0], u1 = bp[2];
      B.u[0] = u0.x; B.u[1] = u0.y; B.u[2] = u0.z; B.u[3] = u0.w;
      B.u[4] = u1.x; B.u[5] = u1.y; B.u[6] = u1.z; B.u[7] = u1.w;
      acc = __builtin_amdgcn_wmma_f32_16x16x32_bf16(
          false, A1[c].v, false, B.v, (short)0, acc, false, false);
    }
    const float wc = Wout[oc];
    #pragma unroll
    for (int r = 0; r < 8; ++r) accout[r] += fmaxf(acc[r], 0.f) * wc;
  }
  // reduce over the 16 columns held across lanes of each half
  #pragma unroll
  for (int m = 1; m <= 8; m <<= 1) {
    #pragma unroll
    for (int r = 0; r < 8; ++r) accout[r] += __shfl_xor(accout[r], m, 32);
  }
  const float bo = bout[0];
  if (mrow < 8) {                       // lanes 0-7 rows 0-7; 16-23 rows 8-15
    float val = accout[0];
    #pragma unroll
    for (int r = 1; r < 8; ++r) val = (mrow == r) ? accout[r] : val;
    const int row = mrow + 8 * half;    // (node_local*4 + kout)
    const int gi  = nodeBase * KOUT + row;
    out[gi] = (val + bo) * (float)out_mask[gi];
  }
}

// Convert W0/W1 fp32 -> bf16 into workspace (64 KB, L2/L0 resident).
__global__ void prep_weights(const float* __restrict__ W0,
                             const float* __restrict__ W1,
                             unsigned short* __restrict__ Wb) {
  const int i = blockIdx.x * blockDim.x + threadIdx.x;
  if (i < H_DIM * H_DIM)            Wb[i] = (unsigned short)(bf_rnd(W0[i]) >> 16);
  else if (i < 2 * H_DIM * H_DIM)   Wb[i] = (unsigned short)(bf_rnd(W1[i - H_DIM * H_DIM]) >> 16);
}

extern "C" void kernel_launch(void* const* d_in, const int* in_sizes, int n_in,
                              void* d_out, int out_size, void* d_ws, size_t ws_size,
                              hipStream_t stream) {
  const float* edges   = (const float*)d_in[0];
  const int*   in_idx  = (const int*)d_in[1];
  const int*   in_mask = (const int*)d_in[2];
  const int*   out_idx = (const int*)d_in[3];
  const int*   out_mask= (const int*)d_in[4];
  const float* W0      = (const float*)d_in[5];
  const float* b0      = (const float*)d_in[6];
  const float* W1      = (const float*)d_in[7];
  const float* b1      = (const float*)d_in[8];
  const float* Wout    = (const float*)d_in[9];
  const float* bout    = (const float*)d_in[10];
  unsigned short* Wb   = (unsigned short*)d_ws;
  float* out           = (float*)d_out;
  const int N = in_sizes[1] / KIN;

  prep_weights<<<(2 * H_DIM * H_DIM + 255) / 256, 256, 0, stream>>>(W0, W1, Wb);
  const int blocks = (N + 31) / 32;   // 32 nodes per block (8 waves x 4)
  encoder_kernel<<<blocks, 256, 0, stream>>>(edges, in_idx, in_mask, out_idx,
                                             out_mask, b0, b1, Wout, bout,
                                             Wb, out, N);
}
